// AttentionBlock_59425167507593
// MI455X (gfx1250) — matrix operand order repaired and checked by reading, not compiled, not run on hardware
//
#include <hip/hip_runtime.h>
#include <hip/hip_bf16.h>
#include <math.h>

// Problem constants (match the JAX reference)
#define B_   2
#define C_   1024
#define T_   2048
#define H_   16
#define G_   32
#define CH_  64
#define CG_  (C_ / G_)   // 32
#define O3_  (3 * C_)    // 3072
#define EPS_ 1e-5f
#define QK_SCALE 0.3535533905932738f  // 1/sqrt(sqrt(CH)) = 8^-0.5

typedef __bf16 bf16;
typedef bf16  v16bf __attribute__((ext_vector_type(16)));
typedef float v8f   __attribute__((ext_vector_type(8)));
typedef int   v4i   __attribute__((vector_size(16)));  // matches builtin param

static __device__ __forceinline__ v8f v8f_zero() {
  v8f z = {0.f, 0.f, 0.f, 0.f, 0.f, 0.f, 0.f, 0.f};
  return z;
}

// D = A(16x32 bf16) * B(32x16 bf16) + C(16x16 f32)
static __device__ __forceinline__ v8f wmma_bf16(v16bf a, v16bf b, v8f c) {
  return __builtin_amdgcn_wmma_f32_16x16x32_bf16(false, a, false, b,
                                                 (short)0, c, false, false);
}

// ---------------------------------------------------------------------------
// Direct global -> LDS 16B tile copy. Prefer the CDNA5 async-to-LDS path
// (tracked by ASYNCcnt, bypasses VGPRs); fall back to a synchronous VGPR
// round-trip if the builtin is unavailable on this toolchain.
// Probe result: builtin exists here with signature
//   (int4 addrspace(1)* src, int4 addrspace(3)* dst, imm int, imm int)
// ---------------------------------------------------------------------------
#if defined(__has_builtin)
#if __has_builtin(__builtin_amdgcn_global_load_async_to_lds_b128)
#define HAS_ASYNC_LDS 1
#endif
#endif

static __device__ __forceinline__ void copy16_g2l(bf16* lds, const bf16* g) {
#if defined(HAS_ASYNC_LDS)
  __builtin_amdgcn_global_load_async_to_lds_b128(
      (__attribute__((address_space(1))) v4i*)(g),
      (__attribute__((address_space(3))) v4i*)(lds), 0, 0);
#else
  *(uint4*)lds = *(const uint4*)g;
#endif
}

static __device__ __forceinline__ void async_wait() {
#if defined(HAS_ASYNC_LDS)
#if __has_builtin(__builtin_amdgcn_s_wait_asynccnt)
  __builtin_amdgcn_s_wait_asynccnt(0);
#else
  asm volatile("s_wait_asynccnt 0x0" ::: "memory");
#endif
#endif
}

// ---------------------------------------------------------------------------
// Kernel 1: GroupNorm. One block per (b, g). Writes xn in f32 (residual path)
// and bf16 (matrix-op operand path).
// ---------------------------------------------------------------------------
__global__ __launch_bounds__(256) void gn_kernel(
    const float* __restrict__ x, const float* __restrict__ scale,
    const float* __restrict__ bias, float* __restrict__ xn,
    bf16* __restrict__ xnb) {
  const int b = blockIdx.x / G_;
  const int g = blockIdx.x % G_;
  const size_t base = ((size_t)b * C_ + (size_t)g * CG_) * T_;
  const int NEL = CG_ * T_;  // 65536
  float s = 0.f, s2 = 0.f;
  for (int i = threadIdx.x; i < NEL; i += 256) {
    float v = x[base + i];
    s += v;
    s2 += v * v;
  }
  __shared__ float sh1[256], sh2[256];
  sh1[threadIdx.x] = s;
  sh2[threadIdx.x] = s2;
  __syncthreads();
  for (int off = 128; off > 0; off >>= 1) {
    if (threadIdx.x < off) {
      sh1[threadIdx.x] += sh1[threadIdx.x + off];
      sh2[threadIdx.x] += sh2[threadIdx.x + off];
    }
    __syncthreads();
  }
  const float mean = sh1[0] / (float)NEL;
  const float var = sh2[0] / (float)NEL - mean * mean;
  const float inv = rsqrtf(var + EPS_);
  for (int i = threadIdx.x; i < NEL; i += 256) {
    int c = g * CG_ + i / T_;
    float v = (x[base + i] - mean) * inv * scale[c] + bias[c];
    xn[base + i] = v;
    xnb[base + i] = (bf16)v;
  }
}

// ---------------------------------------------------------------------------
// Kernel 2: weight conversion f32 -> bf16 (qkv_w and proj_w).
// ---------------------------------------------------------------------------
__global__ __launch_bounds__(256) void cvt_kernel(
    const float* __restrict__ wq, const float* __restrict__ wp,
    bf16* __restrict__ wqb, bf16* __restrict__ wpb) {
  const int stride = gridDim.x * 256;
  const int i0 = blockIdx.x * 256 + threadIdx.x;
  for (int i = i0; i < O3_ * C_; i += stride) wqb[i] = (bf16)wq[i];
  for (int i = i0; i < C_ * C_; i += stride) wpb[i] = (bf16)wp[i];
}

// ---------------------------------------------------------------------------
// Kernel 3/5: tiled WMMA GEMM. Out[z, m, n] = sum_k W[m,k] * X[z, k, n] (+epi)
// 128x128 block tile, BLOCK_K = 64 (two wmma K steps per staged tile).
// 8 waves; each wave owns a 64(m) x 32(n) sub-tile -> 4x2 16x16 accumulators.
// A tile ([m][k], no transpose) is staged with async global->LDS copies; the
// B tile is transposed in-flight through VGPRs (X is [k][n], fragments need
// [n][k]).
// MODE 0: qkv epilogue (bias, fold QK_SCALE into q/k rows, bf16 out)
// MODE 1: proj epilogue (bias + residual, f32 out)
// ---------------------------------------------------------------------------
template <int MODE>
__global__ __launch_bounds__(256) void gemm_kernel(
    const bf16* __restrict__ Wt, const bf16* __restrict__ X,
    const float* __restrict__ bias, const float* __restrict__ resid,
    void* __restrict__ OutP, int Mdim, int Ndim, int Kdim) {
  __shared__ bf16 As[128][80];  // [m][k] 64 K + pad (160B pitch, 16B aligned)
  __shared__ bf16 Bs[128][80];  // [n][k] (X transposed)
  const int tid = threadIdx.x, wid = tid >> 5, lane = tid & 31;
  const int grp = lane >> 4, cn = lane & 15;
  const int wm = wid >> 2, wn = wid & 3;
  const int m0 = blockIdx.y * 128;
  const int n0 = blockIdx.x * 128;
  const size_t xbatch = (size_t)blockIdx.z * (size_t)Kdim * (size_t)Ndim;

  v8f acc[4][2];
#pragma unroll
  for (int i = 0; i < 4; ++i)
#pragma unroll
    for (int j = 0; j < 2; ++j) acc[i][j] = v8f_zero();

  const int tcol = tid & 127;
  const int cb = (tid >> 7) * 32;
  for (int k0 = 0; k0 < Kdim; k0 += 64) {
    // Stage A (W tile, 128x64) with async global->LDS b128 copies.
#pragma unroll
    for (int it = 0; it < 4; ++it) {
      int chunk = tid + it * 256;
      int row = chunk >> 3, cc = (chunk & 7) * 8;
      copy16_g2l(&As[row][cc], Wt + (size_t)(m0 + row) * Kdim + k0 + cc);
    }
    // Stage B (X tile, 64x128) transposed into [n][k]; lanes walk n for
    // coalescing, each thread owns one column and 32 k values. This overlaps
    // with the in-flight async A copies.
    {
      bf16 tmp[32];
#pragma unroll
      for (int c = 0; c < 32; ++c)
        tmp[c] = X[xbatch + (size_t)(k0 + cb + c) * Ndim + n0 + tcol];
#pragma unroll
      for (int q = 0; q < 4; ++q)
        *(uint4*)&Bs[tcol][cb + q * 8] = *(uint4*)&tmp[q * 8];
    }
    if (k0 + 64 < Kdim)  // warm caches for the next X tile
      __builtin_prefetch(X + xbatch + (size_t)(k0 + 64 + cb) * Ndim + n0 + tcol,
                         0, 1);
    async_wait();
    __syncthreads();

#pragma unroll
    for (int kc = 0; kc < 2; ++kc) {
      v16bf af[4], bfg[2];
#pragma unroll
      for (int i = 0; i < 4; ++i) {
        int row = wm * 64 + i * 16 + cn;
        ((uint4*)&af[i])[0] = *(uint4*)&As[row][kc * 32 + grp * 8];
        ((uint4*)&af[i])[1] = *(uint4*)&As[row][kc * 32 + grp * 8 + 16];
      }
#pragma unroll
      for (int j = 0; j < 2; ++j) {
        int col = wn * 32 + j * 16 + cn;
        ((uint4*)&bfg[j])[0] = *(uint4*)&Bs[col][kc * 32 + grp * 8];
        ((uint4*)&bfg[j])[1] = *(uint4*)&Bs[col][kc * 32 + grp * 8 + 16];
      }
#pragma unroll
      for (int i = 0; i < 4; ++i)
#pragma unroll
        for (int j = 0; j < 2; ++j)
          acc[i][j] = wmma_bf16(af[i], bfg[j], acc[i][j]);
    }
    __syncthreads();
  }

  const size_t obatch = (size_t)blockIdx.z * (size_t)Mdim * (size_t)Ndim;
#pragma unroll
  for (int i = 0; i < 4; ++i) {
#pragma unroll
    for (int j = 0; j < 2; ++j) {
      int col = n0 + wn * 32 + j * 16 + cn;
#pragma unroll
      for (int r = 0; r < 8; ++r) {
        int row = m0 + wm * 64 + i * 16 + r + 8 * grp;
        float v = acc[i][j][r] + bias[row];
        if (MODE == 0) {
          // fold attention scale into q and k rows (inner index < 2*CH)
          float sc = ((row % (3 * CH_)) < 2 * CH_) ? QK_SCALE : 1.0f;
          ((bf16*)OutP)[obatch + (size_t)row * Ndim + col] = (bf16)(v * sc);
        } else {
          size_t o = obatch + (size_t)row * Ndim + col;
          ((float*)OutP)[o] = v + resid[o];
        }
      }
    }
  }
}

// ---------------------------------------------------------------------------
// Kernel 4: flash-attention per head. Block = 128 query rows of one (b,head);
// 8 waves x 16 rows. Iterate 64-wide key blocks with online softmax.
// V tile is staged with async global->LDS; qk_bias is staged per-wave into
// LDS with coalesced float4 loads (replaces a per-lane scattered gather).
// ---------------------------------------------------------------------------
__global__ __launch_bounds__(256) void attn_kernel(
    const bf16* __restrict__ qkv, const float* __restrict__ qk_bias,
    bf16* __restrict__ abuf) {
  __shared__ bf16 Qs[128][80];      // [t][c]
  __shared__ bf16 Ks[64][80];       // [s][c]
  __shared__ bf16 Vs[64][80];       // [c][s]
  __shared__ bf16 Ps[8][16][80];    // per-wave P relayout buffer
  __shared__ float BiasS[8][16][64];  // per-wave qk_bias tile
  const int bh = blockIdx.y;
  const int b = bh / H_, head = bh % H_;
  const int t0 = blockIdx.x * 128;
  const bf16* qp = qkv + ((size_t)b * O3_ + (size_t)head * 3 * CH_) * T_;
  const bf16* kp = qp + (size_t)CH_ * T_;
  const bf16* vp = qp + (size_t)(2 * CH_) * T_;
  const int tid = threadIdx.x, wid = tid >> 5, lane = tid & 31;
  const int grp = lane >> 4, cn = lane & 15;
  const int wt0 = t0 + wid * 16;

  // Stage Q transposed (coalesced along t).
  {
    int tcol = tid & 127;
    int cqb = (tid >> 7) * 32;
#pragma unroll 8
    for (int c = 0; c < 32; ++c)
      Qs[tcol][cqb + c] = qp[(size_t)(cqb + c) * T_ + t0 + tcol];
  }
  __syncthreads();

  // Q A-fragments are loop-invariant: load once.
  v16bf qa[2];
  {
    int row = wid * 16 + cn;
#pragma unroll
    for (int kc = 0; kc < 2; ++kc) {
      int base = kc * 32 + grp * 8;
      ((uint4*)&qa[kc])[0] = *(uint4*)&Qs[row][base];
      ((uint4*)&qa[kc])[1] = *(uint4*)&Qs[row][base + 16];
    }
  }

  float mrow[8], lrow[8];
#pragma unroll
  for (int r = 0; r < 8; ++r) {
    mrow[r] = -1e30f;
    lrow[r] = 0.f;
  }
  v8f ov[4];
#pragma unroll
  for (int n = 0; n < 4; ++n) ov[n] = v8f_zero();

  const int sEnd = t0 + 128;  // causal: no keys beyond this block's last row
  for (int s0 = 0; s0 < sEnd; s0 += 64) {
    // Stage V as-is [c][s] with async global->LDS b128 copies.
#pragma unroll
    for (int it = 0; it < 2; ++it) {
      int chunk = tid + it * 256;
      int c = chunk >> 3, sc8 = (chunk & 7) * 8;
      copy16_g2l(&Vs[c][sc8], vp + (size_t)c * T_ + s0 + sc8);
    }
    // Stage K transposed (coalesced along s) while async V is in flight.
    {
      int scol = tid & 63;
      int ckb = (tid >> 6) * 16;
#pragma unroll 8
      for (int c = 0; c < 16; ++c)
        Ks[scol][ckb + c] = kp[(size_t)(ckb + c) * T_ + s0 + scol];
    }
    async_wait();
    __syncthreads();

    if (s0 <= wt0 + 15) {  // wave-level causal block skip (barriers still hit)
      // Per-wave cooperative stage of qk_bias tile (coalesced float4 loads);
      // overlaps with the score WMMAs below.
#pragma unroll
      for (int i = 0; i < 8; ++i) {
        int chunk = lane + 32 * i;  // 0..255
        int rr = chunk >> 4;        // 0..15
        int c4 = (chunk & 15) * 4;  // float4 column
        *(float4*)&BiasS[wid][rr][c4] =
            *(const float4*)(qk_bias + (size_t)(wt0 + rr) * T_ + s0 + c4);
      }
      // scores: 16(t) x 64(s), K = CH = 64 -> 2 wmma K-steps per n-tile
      v8f sacc[4];
#pragma unroll
      for (int n = 0; n < 4; ++n) sacc[n] = v8f_zero();
#pragma unroll
      for (int nt = 0; nt < 4; ++nt) {
#pragma unroll
        for (int kc = 0; kc < 2; ++kc) {
          v16bf kb;
          int scl = nt * 16 + cn;
          ((uint4*)&kb)[0] = *(uint4*)&Ks[scl][kc * 32 + grp * 8];
          ((uint4*)&kb)[1] = *(uint4*)&Ks[scl][kc * 32 + grp * 8 + 16];
          sacc[nt] = wmma_bf16(qa[kc], kb, sacc[nt]);
        }
      }
      // qk_bias + causal mask (C-layout: row = wt0+r+8*grp, col = s0+nt*16+cn)
#pragma unroll
      for (int nt = 0; nt < 4; ++nt) {
        int sg = s0 + nt * 16 + cn;
#pragma unroll
        for (int r = 0; r < 8; ++r) {
          int trow = wt0 + r + 8 * grp;
          float v = sacc[nt][r] + BiasS[wid][r + 8 * grp][nt * 16 + cn];
          sacc[nt][r] = (sg <= trow) ? v : -1e30f;
        }
      }
      // online softmax; each 16-lane half-wave shares one set of 8 rows
#pragma unroll
      for (int r = 0; r < 8; ++r) {
        float mx = sacc[0][r];
#pragma unroll
        for (int nt = 1; nt < 4; ++nt) mx = fmaxf(mx, sacc[nt][r]);
#pragma unroll
        for (int m = 1; m < 16; m <<= 1) mx = fmaxf(mx, __shfl_xor(mx, m));
        float mnew = fmaxf(mrow[r], mx);
        float alpha = __expf(mrow[r] - mnew);
        float ps = 0.f;
#pragma unroll
        for (int nt = 0; nt < 4; ++nt) {
          float p = __expf(sacc[nt][r] - mnew);
          sacc[nt][r] = p;
          ps += p;
        }
#pragma unroll
        for (int m = 1; m < 16; m <<= 1) ps += __shfl_xor(ps, m);
        lrow[r] = lrow[r] * alpha + ps;
        mrow[r] = mnew;
#pragma unroll
        for (int n = 0; n < 4; ++n) ov[n][r] *= alpha;
      }
      // relayout P (C-layout -> A-fragment layout) through per-wave LDS
#pragma unroll
      for (int nt = 0; nt < 4; ++nt)
#pragma unroll
        for (int r = 0; r < 8; ++r)
          Ps[wid][r + 8 * grp][nt * 16 + cn] = (bf16)sacc[nt][r];
      v16bf pa[2];
#pragma unroll
      for (int kc = 0; kc < 2; ++kc) {
        int base = kc * 32 + grp * 8;
        ((uint4*)&pa[kc])[0] = *(uint4*)&Ps[wid][cn][base];
        ((uint4*)&pa[kc])[1] = *(uint4*)&Ps[wid][cn][base + 16];
      }
      // O += P * V^T   (B fragment reads Vs[c][s] directly)
#pragma unroll
      for (int ntc = 0; ntc < 4; ++ntc) {
#pragma unroll
        for (int kc = 0; kc < 2; ++kc) {
          v16bf vb;
          int cl = ntc * 16 + cn;
          ((uint4*)&vb)[0] = *(uint4*)&Vs[cl][kc * 32 + grp * 8];
          ((uint4*)&vb)[1] = *(uint4*)&Vs[cl][kc * 32 + grp * 8 + 16];
          ov[ntc] = wmma_bf16(pa[kc], vb, ov[ntc]);
        }
      }
    }
    __syncthreads();
  }

  // normalize and scatter a[t,c] -> abuf[b, head*CH + c, t] (bf16)
#pragma unroll
  for (int ntc = 0; ntc < 4; ++ntc) {
    int cl = ntc * 16 + cn;
#pragma unroll
    for (int r = 0; r < 8; ++r) {
      int trow = wt0 + r + 8 * grp;
      float o = ov[ntc][r] / lrow[r];
      abuf[((size_t)b * C_ + (size_t)head * CH_ + cl) * T_ + trow] = (bf16)o;
    }
  }
}

// ---------------------------------------------------------------------------
// Host launcher. Inputs: x, mask(unused; causal computed), qk_bias, gn_scale,
// gn_bias, qkv_w, qkv_b, proj_w, proj_b. Output: f32 (B,C,T).
// ---------------------------------------------------------------------------
extern "C" void kernel_launch(void* const* d_in, const int* in_sizes, int n_in,
                              void* d_out, int out_size, void* d_ws,
                              size_t ws_size, hipStream_t stream) {
  (void)in_sizes; (void)n_in; (void)out_size; (void)ws_size;
  const float* x        = (const float*)d_in[0];
  const float* qk_bias  = (const float*)d_in[2];
  const float* gn_scale = (const float*)d_in[3];
  const float* gn_bias  = (const float*)d_in[4];
  const float* qkv_w    = (const float*)d_in[5];
  const float* qkv_b    = (const float*)d_in[6];
  const float* proj_w   = (const float*)d_in[7];
  const float* proj_b   = (const float*)d_in[8];

  char* ws = (char*)d_ws;
  size_t off = 0;
  float* xn_f32 = (float*)(ws + off); off += (size_t)B_ * C_ * T_ * 4;
  bf16* xn_b    = (bf16*)(ws + off);  off += (size_t)B_ * C_ * T_ * 2;
  bf16* wq_b    = (bf16*)(ws + off);  off += (size_t)O3_ * C_ * 2;
  bf16* wp_b    = (bf16*)(ws + off);  off += (size_t)C_ * C_ * 2;
  bf16* qkv_buf = (bf16*)(ws + off);  off += (size_t)B_ * O3_ * T_ * 2;
  bf16* a_buf   = (bf16*)(ws + off);  off += (size_t)B_ * C_ * T_ * 2;

  gn_kernel<<<dim3(B_ * G_), dim3(256), 0, stream>>>(x, gn_scale, gn_bias,
                                                     xn_f32, xn_b);
  cvt_kernel<<<dim3(2048), dim3(256), 0, stream>>>(qkv_w, proj_w, wq_b, wp_b);
  gemm_kernel<0><<<dim3(T_ / 128, O3_ / 128, B_), dim3(256), 0, stream>>>(
      wq_b, xn_b, qkv_b, nullptr, (void*)qkv_buf, O3_, T_, C_);
  attn_kernel<<<dim3(T_ / 128, B_ * H_), dim3(256), 0, stream>>>(
      qkv_buf, qk_bias, a_buf);
  gemm_kernel<1><<<dim3(T_ / 128, C_ / 128, B_), dim3(256), 0, stream>>>(
      wp_b, a_buf, proj_b, xn_f32, d_out, C_, T_, C_);
}
